// RSNN_Model_80882824118449
// MI455X (gfx1250) — compile-verified, matrix operand order checked by reading, not run on hardware
//
#include <hip/hip_runtime.h>
#include <hip/hip_bf16.h>

// ---- problem sizes from the reference ----
#define B_      1024
#define WIN_    25
#define NIN_    2312            // 34*34*2 ; = 72*32 + 8
#define NH_     1024
#define NOUT_   10
#define MROWS_  (B_ * WIN_)     // 25600
#define THRESH_ 0.3f
#define KFULL_  2304            // 72 full K-steps of 32
#define NIT_    72

typedef __bf16 bf16_t;
typedef __attribute__((ext_vector_type(16))) __bf16 bf16x16;
typedef __attribute__((ext_vector_type(8)))  __bf16 bf16x8;
typedef __attribute__((ext_vector_type(4)))  __bf16 bf16x4;
typedef __attribute__((ext_vector_type(8)))  float  f32x8;
typedef __attribute__((ext_vector_type(4)))  float  f32x4;

// 16-element bf16 fragment = two contiguous 16-byte chunks
// (ISA 7.12.2: 16-bit 16x32 layout -> per lane K {base..base+7} U {base+16..base+23})
static __device__ __forceinline__ bf16x16 frag_ld(const bf16_t* lo, const bf16_t* hi) {
  union { bf16x16 v; bf16x8 h[2]; } u;
  u.h[0] = *(const bf16x8*)lo;
  u.h[1] = *(const bf16x8*)hi;
  return u.v;
}

static __device__ __forceinline__ f32x8 wmma_bf16(bf16x16 a, bf16x16 b, f32x8 c) {
  // (neg_a, A, neg_b, B, c_mod, C, reuse_a, reuse_b)
  return __builtin_amdgcn_wmma_f32_16x16x32_bf16(false, a, false, b, (short)0, c, false, false);
}

// ============ phase 0: weight conversion to bf16 ============
__global__ void k_cvt_bf16(const float* __restrict__ s, bf16_t* __restrict__ d, int n) {
  int i = blockIdx.x * blockDim.x + threadIdx.x;
  if (i < n) d[i] = (bf16_t)s[i];
}

// w_ho (10x1024) -> bf16 padded to 16x1024 (rows 10..15 = 0)
__global__ void k_pad_who(const float* __restrict__ w, bf16_t* __restrict__ d) {
  int i = blockIdx.x * blockDim.x + threadIdx.x;   // 16*1024 threads
  int r = i >> 10, c = i & 1023;
  float v = (r < NOUT_) ? w[r * NH_ + c] : 0.0f;
  d[i] = (bf16_t)v;
}

// ============ phase 1: pre[r, n] = sum_k inp[r, k] * w_ih[n, k] ============
// M=25600, N=1024, K=2312.  Block tile 32x256, K-step 32, 256 threads (8 waves).
// Each wave computes 4 C tiles (2 M-subtiles x 2 N-subtiles) -> A/B frags reused 2x.
// 2-stage pipeline: next global tile loads issue before this tile's WMMAs.
__global__ __launch_bounds__(256) void k_pre_gemm(const float* __restrict__ inp,
                                                  const bf16_t* __restrict__ wih,
                                                  float* __restrict__ pre) {
  __shared__ bf16_t lA[32 * 32];    // A tile  [m][k]
  __shared__ bf16_t lB[256 * 32];   // B tile  [n][k]
  const int tid  = threadIdx.x;
  const int lane = tid & 31;
  const int wave = tid >> 5;
  const int m0 = blockIdx.x * 32;    // 800 blocks
  const int n0 = blockIdx.y * 256;   // 4 blocks
  const int half8 = (lane < 16) ? 0 : 8;
  const int l15   = lane & 15;

  // per-thread staging assignments
  const int ar  = tid >> 3;            // A row (0..31)
  const int ac4 = (tid & 7) * 4;       // A k-chunk of 4
  f32x4  ra;                           // A stage (f32)
  bf16x8 rb[4];                        // B stage (4 chunks of 8 bf16)

  // prologue: stage tile kk=0
  ra = *(const f32x4*)(inp + (size_t)(m0 + ar) * NIN_ + ac4);
  #pragma unroll
  for (int e = 0; e < 4; ++e) {
    int idx = e * 256 + tid, n = idx >> 2, c = idx & 3;
    rb[e] = *(const bf16x8*)(wih + (size_t)(n0 + n) * NIN_ + c * 8);
  }

  f32x8 acc[4] = {};
  const bf16_t* ap0 = lA + l15 * 32 + half8;
  const bf16_t* ap1 = lA + (16 + l15) * 32 + half8;
  const bf16_t* bp0 = lB + (wave * 32 + l15) * 32 + half8;
  const bf16_t* bp1 = lB + (wave * 32 + 16 + l15) * 32 + half8;

  for (int it = 0; it < NIT_; ++it) {
    __syncthreads();                       // LDS free (prev frag reads done)
    // commit staged tile to LDS
    {
      bf16x4 v;
      #pragma unroll
      for (int q = 0; q < 4; ++q) v[q] = (bf16_t)ra[q];
      *(bf16x4*)(lA + ar * 32 + ac4) = v;
    }
    #pragma unroll
    for (int e = 0; e < 4; ++e) {
      int idx = e * 256 + tid, n = idx >> 2, c = idx & 3;
      *(bf16x8*)(lB + n * 32 + c * 8) = rb[e];
    }
    __syncthreads();                       // tile visible
    // stage next tile (overlaps with frag loads + WMMA below)
    if (it + 1 < NIT_) {
      int kk = (it + 1) * 32;
      ra = *(const f32x4*)(inp + (size_t)(m0 + ar) * NIN_ + kk + ac4);
      #pragma unroll
      for (int e = 0; e < 4; ++e) {
        int idx = e * 256 + tid, n = idx >> 2, c = idx & 3;
        rb[e] = *(const bf16x8*)(wih + (size_t)(n0 + n) * NIN_ + kk + c * 8);
      }
    }
    // compute: 2x2 tiles, A/B fragments reused twice each
    bf16x16 a0 = frag_ld(ap0, ap0 + 16);
    bf16x16 a1 = frag_ld(ap1, ap1 + 16);
    bf16x16 b0 = frag_ld(bp0, bp0 + 16);
    bf16x16 b1 = frag_ld(bp1, bp1 + 16);
    acc[0] = wmma_bf16(a0, b0, acc[0]);
    acc[1] = wmma_bf16(a0, b1, acc[1]);
    acc[2] = wmma_bf16(a1, b0, acc[2]);
    acc[3] = wmma_bf16(a1, b1, acc[3]);
  }

  // ---- peeled K tail: kk = 2304, 8 valid k, zero-padded ----
  __syncthreads();
  {
    bf16x4 v = {};
    #pragma unroll
    for (int q = 0; q < 4; ++q) {
      int k = KFULL_ + ac4 + q;
      if (k < NIN_) v[q] = (bf16_t)inp[(size_t)(m0 + ar) * NIN_ + k];
    }
    *(bf16x4*)(lA + ar * 32 + ac4) = v;
  }
  #pragma unroll
  for (int e = 0; e < 4; ++e) {
    int idx = e * 256 + tid, n = idx >> 2, c = idx & 3;
    bf16x8 v = {};
    if (c == 0) v = *(const bf16x8*)(wih + (size_t)(n0 + n) * NIN_ + KFULL_);
    *(bf16x8*)(lB + n * 32 + c * 8) = v;
  }
  __syncthreads();
  {
    bf16x16 a0 = frag_ld(ap0, ap0 + 16);
    bf16x16 a1 = frag_ld(ap1, ap1 + 16);
    bf16x16 b0 = frag_ld(bp0, bp0 + 16);
    bf16x16 b1 = frag_ld(bp1, bp1 + 16);
    acc[0] = wmma_bf16(a0, b0, acc[0]);
    acc[1] = wmma_bf16(a0, b1, acc[1]);
    acc[2] = wmma_bf16(a1, b0, acc[2]);
    acc[3] = wmma_bf16(a1, b1, acc[3]);
  }

  // C layout: vgpr i -> row half8+i, col l15
  #pragma unroll
  for (int mh = 0; mh < 2; ++mh)
    #pragma unroll
    for (int nt = 0; nt < 2; ++nt)
      #pragma unroll
      for (int i = 0; i < 8; ++i)
        pre[(size_t)(m0 + mh * 16 + half8 + i) * NH_ +
            n0 + wave * 32 + nt * 16 + l15] = acc[mh * 2 + nt][i];
}

// ============ phase 2: persistent recurrent LIF, 16 batch rows per WG ============
// 8 waves; wave w owns hidden columns [w*128, w*128+128) as 8 WMMA tiles.
// Spikes (16x1024 bf16) in LDS = WMMA A operand. w_hh/w_ho bf16 stream from L2,
// with global_prefetch of the next K-block overlapping the current WMMA burst.
__global__ __launch_bounds__(256) void k_rsnn_recur(const float* __restrict__ pre,
                                                     const bf16_t* __restrict__ whh,
                                                     const bf16_t* __restrict__ who,
                                                     float* __restrict__ out) {
  __shared__ bf16_t spk[16 * NH_];   // 32 KB
  const int tid  = threadIdx.x;
  const int lane = tid & 31;
  const int wave = tid >> 5;
  const int b0   = blockIdx.x * 16;
  const int half8 = (lane < 16) ? 0 : 8;
  const int l15   = lane & 15;
  const float alpha = expf(-1.0f / 9.4912f);

  // initial state: spikes = 0, membranes = 0
  for (int i = tid; i < 16 * NH_; i += 256) spk[i] = (bf16_t)0.0f;
  f32x8 hm[8] = {};                 // reset membrane (row half8+i, col wave*128+j*16+l15)
  f32x8 omem = {}, osum = {};       // used by wave 0 only
  __syncthreads();

  for (int t = 0; t < WIN_; ++t) {
    // ---- hidden GEMM: acc = spikes(prev) @ w_hh^T over this wave's 128 cols ----
    f32x8 acc[8] = {};
    for (int kk = 0; kk < NH_; kk += 32) {
      const bf16_t* ap = spk + l15 * NH_ + kk + half8;
      bf16x16 a = frag_ld(ap, ap + 16);          // shared across the 8 N-tiles
      if (kk + 32 < NH_) {                       // prefetch next K-block (L2 -> near)
        #pragma unroll
        for (int j = 0; j < 8; ++j) {
          int n = wave * 128 + j * 16 + l15;
          __builtin_prefetch(whh + (size_t)n * NH_ + kk + 32 + half8, 0, 3);
        }
      }
      #pragma unroll
      for (int j = 0; j < 8; ++j) {
        int n = wave * 128 + j * 16 + l15;       // B col n = w_hh row n
        const bf16_t* bp = whh + (size_t)n * NH_ + kk + half8;
        bf16x16 b = frag_ld(bp, bp + 16);
        acc[j] = wmma_bf16(a, b, acc[j]);
      }
    }
    __syncthreads();   // everyone (incl. wave0's prev output GEMM) done reading old spikes
    // ---- LIF update + spike write (reset makes the (1-spike) factor a no-op) ----
    #pragma unroll
    for (int j = 0; j < 8; ++j) {
      int ncol = wave * 128 + j * 16 + l15;
      #pragma unroll
      for (int i = 0; i < 8; ++i) {
        int mrow = half8 + i;
        float p = pre[((size_t)(b0 + mrow) * WIN_ + t) * NH_ + ncol];
        float v = p + acc[j][i] + alpha * hm[j][i];
        spk[mrow * NH_ + ncol] = (bf16_t)((v > THRESH_) ? 1.0f : 0.0f);
        hm[j][i] = (v < THRESH_) ? v : 0.0f;
      }
    }
    __syncthreads();   // new spikes visible
    // ---- output layer (wave 0, full-wave EXEC): 16x16 tile, cols 10..15 padded 0 ----
    if (wave == 0) {
      f32x8 oacc = {};
      for (int kk = 0; kk < NH_; kk += 32) {
        const bf16_t* ap = spk + l15 * NH_ + kk + half8;
        const bf16_t* bp = who + (size_t)l15 * NH_ + kk + half8;
        oacc = wmma_bf16(frag_ld(ap, ap + 16), frag_ld(bp, bp + 16), oacc);
      }
      #pragma unroll
      for (int i = 0; i < 8; ++i) {
        float v = alpha * omem[i] + oacc[i];
        osum[i] += (v > THRESH_) ? 1.0f : 0.0f;
        omem[i] = (v < THRESH_) ? v : 0.0f;
      }
    }
  }
  if (wave == 0 && l15 < NOUT_) {
    #pragma unroll
    for (int i = 0; i < 8; ++i)
      out[(size_t)(b0 + half8 + i) * NOUT_ + l15] = osum[i] / (float)WIN_;
  }
}

// ============ host-side launch ============
extern "C" void kernel_launch(void* const* d_in, const int* in_sizes, int n_in,
                              void* d_out, int out_size, void* d_ws, size_t ws_size,
                              hipStream_t stream) {
  const float* inp  = (const float*)d_in[0];   // [B, WIN, NIN]
  const float* w_ih = (const float*)d_in[1];   // [NH, NIN]
  const float* w_hh = (const float*)d_in[2];   // [NH, NH]
  const float* w_ho = (const float*)d_in[3];   // [NOUT, NH]
  float* out = (float*)d_out;                  // [B, NOUT]

  // workspace layout (16B-aligned offsets)
  char* ws = (char*)d_ws;
  float*  pre    = (float*)ws;                                   // 25600*1024*4 = 104857600 B
  bf16_t* wih_bf = (bf16_t*)(ws + (size_t)104857600);            // 1024*2312*2  =   4734976 B
  bf16_t* whh_bf = (bf16_t*)(ws + (size_t)104857600 + 4734976);  // 1024*1024*2  =   2097152 B
  bf16_t* who_bf = (bf16_t*)(ws + (size_t)104857600 + 4734976 + 2097152); // 16*1024*2

  // phase 0: conversions
  {
    int n = NH_ * NIN_;
    k_cvt_bf16<<<(n + 255) / 256, 256, 0, stream>>>(w_ih, wih_bf, n);
  }
  {
    int n = NH_ * NH_;
    k_cvt_bf16<<<(n + 255) / 256, 256, 0, stream>>>(w_hh, whh_bf, n);
  }
  k_pad_who<<<(16 * NH_) / 256, 256, 0, stream>>>(w_ho, who_bf);

  // phase 1: input projection GEMM (25600 x 1024 x 2312)
  k_pre_gemm<<<dim3(MROWS_ / 32, NH_ / 256), 256, 0, stream>>>(inp, wih_bf, pre);

  // phase 2: persistent batch-parallel recurrence (64 WGs x 25 steps)
  k_rsnn_recur<<<B_ / 16, 256, 0, stream>>>(pre, whh_bf, who_bf, out);
}
